// InferDecoder_28552942583889
// MI455X (gfx1250) — matrix operand order repaired and checked by reading, not compiled, not run on hardware
//
#include <hip/hip_runtime.h>
#include <hip/hip_bf16.h>
#include <math.h>

// ---------------- problem constants ----------------
#define B_    32
#define T_    128
#define V_    32000
#define E_    512
#define H_    1024
#define G4H_  4096
#define START_TOK 1

typedef __attribute__((ext_vector_type(16))) __bf16 v16bf;
typedef __attribute__((ext_vector_type(8)))  __bf16 v8bf;
typedef __attribute__((ext_vector_type(8)))  float  v8f;

// ---------------------------------------------------------------------------
// One-time conversion: Wt[n][k] = bf16( k < inDim ? Wx[k][n] : Wh[k-inDim][n] )
// Layout [4096][Kdim] row-major => WMMA-B fragments are contiguous 32B loads.
// ---------------------------------------------------------------------------
__global__ void k_convert_w(const float* __restrict__ Wx,
                            const float* __restrict__ Wh,
                            __bf16* __restrict__ Wt,
                            int inDim, int Kdim) {
    int idx = blockIdx.x * blockDim.x + threadIdx.x;
    int total = G4H_ * Kdim;
    if (idx >= total) return;
    int n = idx / Kdim;
    int k = idx - n * Kdim;
    float v = (k < inDim) ? Wx[(size_t)k * G4H_ + n]
                          : Wh[(size_t)(k - inDim) * G4H_ + n];
    Wt[idx] = (__bf16)v;
}

// fcWt[v][k] = bf16(fcW[k][v])   ([32000][1024] row-major)
__global__ void k_convert_fc(const float* __restrict__ fcW,
                             __bf16* __restrict__ fcWt) {
    int idx = blockIdx.x * blockDim.x + threadIdx.x;
    if (idx >= V_ * H_) return;
    int v = idx / H_;
    int k = idx - v * H_;
    fcWt[idx] = (__bf16)fcW[(size_t)k * V_ + v];
}

// copy initial h/c states into workspace, init tokens to START
__global__ void k_init(const float* h0, const float* c0,
                       const float* h1, const float* c1,
                       const float* h2, const float* c2,
                       const float* h3, const float* c3,
                       float* __restrict__ hs, float* __restrict__ cs,
                       int* __restrict__ tok) {
    int gid = blockIdx.x * blockDim.x + threadIdx.x;
    const int per = B_ * H_;                  // 32768 per layer
    if (gid < 4 * per) {
        int l = gid / per;
        int r = gid - l * per;
        const float* hsrc = (l == 0) ? h0 : (l == 1) ? h1 : (l == 2) ? h2 : h3;
        const float* csrc = (l == 0) ? c0 : (l == 1) ? c1 : (l == 2) ? c2 : c3;
        hs[gid] = hsrc[r];
        cs[gid] = csrc[r];
    }
    if (gid < B_) tok[gid] = START_TOK;
}

// Build layer-0 A operand: A0[b][0:512]=emb[tok[b]], A0[b][512:1536]=h0 (bf16)
__global__ void k_prepare0(const float* __restrict__ emb,
                           const float* __restrict__ hs0,
                           const int* __restrict__ tok,
                           __bf16* __restrict__ A0) {
    int idx = blockIdx.x * blockDim.x + threadIdx.x;  // 32*1536
    if (idx >= B_ * (E_ + H_)) return;
    int b = idx / (E_ + H_);
    int col = idx - b * (E_ + H_);
    float v = (col < E_) ? emb[(size_t)tok[b] * E_ + col]
                         : hs0[b * H_ + (col - E_)];
    A0[idx] = (__bf16)v;
}

__device__ __forceinline__ float sigf(float x) { return 1.0f / (1.0f + expf(-x)); }

// ---------------------------------------------------------------------------
// Fused LSTM layer:  z = A @ Wt^T + b ; gates ; c,h update ; build next A.
//   A     : [32][Kdim] bf16 (x ‖ h_prev)
//   Wt    : [4096][Kdim] bf16 (pre-transposed, gate g at rows g*1024..)
//   grid  : 16 blocks x 256 thr (8 waves); block covers j in [blk*64, blk*64+64)
//   wave  : m_tile = w>>2 (batch 0-15 / 16-31), n_sub = w&3 (16 cols)
//   4 accumulators per wave: one per gate (i,f,g,o) -> full fusion in-register
// ---------------------------------------------------------------------------
__global__ void __launch_bounds__(256)
k_lstm(const __bf16* __restrict__ A, const __bf16* __restrict__ Wt,
       const float* __restrict__ bias,
       float* __restrict__ hs, float* __restrict__ cs,
       const float* __restrict__ h_next,     // next layer's h state (or null)
       __bf16* __restrict__ Anext, int Kdim, int AnextK) {
    const int lane   = threadIdx.x & 31;
    const int wave   = threadIdx.x >> 5;
    const int m_tile = wave >> 2;
    const int n_sub  = wave & 3;
    const int j0     = blockIdx.x * 64 + n_sub * 16;
    const int hi     = lane >> 4;
    const int l15    = lane & 15;

    v8f acc_i = {}, acc_f = {}, acc_g = {}, acc_o = {};

    const __bf16* Arow = A + (size_t)(m_tile * 16 + l15) * Kdim;
    const int colB = j0 + l15;
    const __bf16* Bi = Wt + (size_t)(0 * H_ + colB) * Kdim + hi * 16;
    const __bf16* Bf = Wt + (size_t)(1 * H_ + colB) * Kdim + hi * 16;
    const __bf16* Bg = Wt + (size_t)(2 * H_ + colB) * Kdim + hi * 16;
    const __bf16* Bo = Wt + (size_t)(3 * H_ + colB) * Kdim + hi * 16;

    for (int k = 0; k < Kdim; k += 32) {
        // A fragment: lane holds row m, K = k+{0..15} (+16 if hi half)
        v8bf alo = *(const v8bf*)(Arow + k + hi * 16);
        v8bf ahi = *(const v8bf*)(Arow + k + 8 + hi * 16);
        v16bf a;
#pragma unroll
        for (int i = 0; i < 8; ++i) { a[i] = alo[i]; a[i + 8] = ahi[i]; }

        // B fragments: 16 contiguous bf16 per lane (column colB, K-half by hi)
        v16bf bi = *(const v16bf*)(Bi + k);
        v16bf bf = *(const v16bf*)(Bf + k);
        v16bf bg = *(const v16bf*)(Bg + k);
        v16bf bo = *(const v16bf*)(Bo + k);
        __builtin_prefetch(Bi + k + 512, 0, 1);

        acc_i = __builtin_amdgcn_wmma_f32_16x16x32_bf16(false, a, false, bi, (short)0, acc_i, false, false);
        acc_f = __builtin_amdgcn_wmma_f32_16x16x32_bf16(false, a, false, bf, (short)0, acc_f, false, false);
        acc_g = __builtin_amdgcn_wmma_f32_16x16x32_bf16(false, a, false, bg, (short)0, acc_g, false, false);
        acc_o = __builtin_amdgcn_wmma_f32_16x16x32_bf16(false, a, false, bo, (short)0, acc_o, false, false);
    }

    // C/D layout: VGPR r, lanes<16 -> row r, lanes>=16 -> row r+8; col = lane&15
    const int j = j0 + l15;
#pragma unroll
    for (int r = 0; r < 8; ++r) {
        int b = m_tile * 16 + r + hi * 8;
        float zi = acc_i[r] + bias[j];
        float zf = acc_f[r] + bias[H_ + j];
        float zg = acc_g[r] + bias[2 * H_ + j];
        float zo = acc_o[r] + bias[3 * H_ + j];
        float ig = sigf(zi);
        float fg = sigf(zf);
        float gg = tanhf(zg);
        float og = sigf(zo);
        float cn = fg * cs[b * H_ + j] + ig * gg;
        float hn = og * tanhf(cn);
        cs[b * H_ + j] = cn;
        hs[b * H_ + j] = hn;
        Anext[(size_t)b * AnextK + j] = (__bf16)hn;
        if (h_next)  // concat next layer's previous h into its A operand
            Anext[(size_t)b * AnextK + H_ + j] = (__bf16)h_next[b * H_ + j];
    }
}

// ---------------------------------------------------------------------------
// FC: logits[b][v] = h3 @ fcWt^T + fcb, written directly to d_out[b][t][v].
// grid 250 x 256thr; wave handles one 16-col tile, both batch halves.
// ---------------------------------------------------------------------------
__global__ void __launch_bounds__(256)
k_fc(const __bf16* __restrict__ Afc, const __bf16* __restrict__ fcWt,
     const float* __restrict__ fcb, float* __restrict__ out, int t) {
    const int lane = threadIdx.x & 31;
    const int wave = threadIdx.x >> 5;
    const int hi   = lane >> 4;
    const int l15  = lane & 15;
    const int v0   = (blockIdx.x * 8 + wave) * 16;

    v8f acc0 = {}, acc1 = {};
    const __bf16* Ar0 = Afc + (size_t)l15 * H_;
    const __bf16* Ar1 = Afc + (size_t)(16 + l15) * H_;
    const __bf16* Bp  = fcWt + (size_t)(v0 + l15) * H_ + hi * 16;

    for (int k = 0; k < H_; k += 32) {
        v8bf a0lo = *(const v8bf*)(Ar0 + k + hi * 16);
        v8bf a0hi = *(const v8bf*)(Ar0 + k + 8 + hi * 16);
        v8bf a1lo = *(const v8bf*)(Ar1 + k + hi * 16);
        v8bf a1hi = *(const v8bf*)(Ar1 + k + 8 + hi * 16);
        v16bf a0, a1;
#pragma unroll
        for (int i = 0; i < 8; ++i) {
            a0[i] = a0lo[i]; a0[i + 8] = a0hi[i];
            a1[i] = a1lo[i]; a1[i + 8] = a1hi[i];
        }
        v16bf b = *(const v16bf*)(Bp + k);
        __builtin_prefetch(Bp + k + 512, 0, 1);
        acc0 = __builtin_amdgcn_wmma_f32_16x16x32_bf16(false, a0, false, b, (short)0, acc0, false, false);
        acc1 = __builtin_amdgcn_wmma_f32_16x16x32_bf16(false, a1, false, b, (short)0, acc1, false, false);
    }

    const int v = v0 + l15;
    const float bb = fcb[v];
#pragma unroll
    for (int r = 0; r < 8; ++r) {
        int b0 = r + hi * 8;
        out[((size_t)b0 * T_ + t) * V_ + v]        = acc0[r] + bb;
        out[((size_t)(b0 + 16) * T_ + t) * V_ + v] = acc1[r] + bb;
    }
}

// ---------------------------------------------------------------------------
// Argmax over V with first-occurrence tie-break (== jnp.argmax).
// One block per batch row.
// ---------------------------------------------------------------------------
__global__ void k_argmax(const float* __restrict__ out, int t,
                         int* __restrict__ tok, float* __restrict__ tokout) {
    const int b = blockIdx.x;
    const int tid = threadIdx.x;
    const float* row = out + ((size_t)b * T_ + t) * V_;

    float best = -3.402823466e38f;
    int   bi   = 0x7FFFFFFF;
    for (int v = tid; v < V_; v += 256) {
        float x = row[v];
        if (x > best || (x == best && v < bi)) { best = x; bi = v; }
    }
    __shared__ float sv[256];
    __shared__ int   si[256];
    sv[tid] = best; si[tid] = bi;
    __syncthreads();
    for (int s = 128; s > 0; s >>= 1) {
        if (tid < s) {
            if (sv[tid + s] > sv[tid] ||
                (sv[tid + s] == sv[tid] && si[tid + s] < si[tid])) {
                sv[tid] = sv[tid + s]; si[tid] = si[tid + s];
            }
        }
        __syncthreads();
    }
    if (tid == 0) {
        tok[b] = si[0];
        tokout[(size_t)b * T_ + t] = (float)si[0];
    }
}

// ---------------------------------------------------------------------------
extern "C" void kernel_launch(void* const* d_in, const int* in_sizes, int n_in,
                              void* d_out, int out_size, void* d_ws, size_t ws_size,
                              hipStream_t stream) {
    (void)in_sizes; (void)n_in; (void)out_size; (void)ws_size;
    const float* h_in[4] = {(const float*)d_in[0], (const float*)d_in[2],
                            (const float*)d_in[4], (const float*)d_in[6]};
    const float* c_in[4] = {(const float*)d_in[1], (const float*)d_in[3],
                            (const float*)d_in[5], (const float*)d_in[7]};
    const float* emb = (const float*)d_in[8];
    const float* Wx[4]   = {(const float*)d_in[9],  (const float*)d_in[12],
                            (const float*)d_in[15], (const float*)d_in[18]};
    const float* Wh[4]   = {(const float*)d_in[10], (const float*)d_in[13],
                            (const float*)d_in[16], (const float*)d_in[19]};
    const float* bias[4] = {(const float*)d_in[11], (const float*)d_in[14],
                            (const float*)d_in[17], (const float*)d_in[20]};
    const float* fcW = (const float*)d_in[21];
    const float* fcb = (const float*)d_in[22];

    // -------- workspace layout (~130 MB; fits entirely in the 192MB L2) -----
    char* ws = (char*)d_ws;
    size_t off = 0;
    auto walloc = [&](size_t bytes) -> char* {
        char* p = ws + off;
        off += (bytes + 255) & ~(size_t)255;
        return p;
    };
    __bf16* Wt[4];
    Wt[0] = (__bf16*)walloc((size_t)G4H_ * 1536 * 2);
    for (int l = 1; l < 4; ++l) Wt[l] = (__bf16*)walloc((size_t)G4H_ * 2048 * 2);
    __bf16* fcWt = (__bf16*)walloc((size_t)V_ * H_ * 2);
    float*  hs   = (float*)walloc((size_t)4 * B_ * H_ * 4);
    float*  cs   = (float*)walloc((size_t)4 * B_ * H_ * 4);
    __bf16* A0   = (__bf16*)walloc((size_t)B_ * 1536 * 2);
    __bf16* A1   = (__bf16*)walloc((size_t)B_ * 2048 * 2);
    __bf16* A2   = (__bf16*)walloc((size_t)B_ * 2048 * 2);
    __bf16* A3   = (__bf16*)walloc((size_t)B_ * 2048 * 2);
    __bf16* Afc  = (__bf16*)walloc((size_t)B_ * H_ * 2);
    int*    tok  = (int*)walloc(B_ * 4);

    // -------- one-time per launch: bf16 transposed weights + state init -----
    {
        int total = G4H_ * 1536;
        k_convert_w<<<(total + 255) / 256, 256, 0, stream>>>(Wx[0], Wh[0], Wt[0], E_, 1536);
    }
    for (int l = 1; l < 4; ++l) {
        int total = G4H_ * 2048;
        k_convert_w<<<(total + 255) / 256, 256, 0, stream>>>(Wx[l], Wh[l], Wt[l], H_, 2048);
    }
    {
        int total = V_ * H_;
        k_convert_fc<<<(total + 255) / 256, 256, 0, stream>>>(fcW, fcWt);
    }
    k_init<<<(4 * B_ * H_ + 255) / 256, 256, 0, stream>>>(
        h_in[0], c_in[0], h_in[1], c_in[1], h_in[2], c_in[2], h_in[3], c_in[3],
        hs, cs, tok);

    float* logits = (float*)d_out;
    float* tokout = (float*)d_out + (size_t)B_ * T_ * V_;
    const int SPL = B_ * H_;  // state stride per layer

    // -------- 128 serialized decode steps -----------------------------------
    for (int t = 0; t < T_; ++t) {
        k_prepare0<<<(B_ * 1536 + 255) / 256, 256, 0, stream>>>(emb, hs, tok, A0);
        k_lstm<<<16, 256, 0, stream>>>(A0, Wt[0], bias[0], hs,           cs,
                                       hs + 1 * SPL, A1, 1536, 2048);
        k_lstm<<<16, 256, 0, stream>>>(A1, Wt[1], bias[1], hs + 1 * SPL, cs + 1 * SPL,
                                       hs + 2 * SPL, A2, 2048, 2048);
        k_lstm<<<16, 256, 0, stream>>>(A2, Wt[2], bias[2], hs + 2 * SPL, cs + 2 * SPL,
                                       hs + 3 * SPL, A3, 2048, 2048);
        k_lstm<<<16, 256, 0, stream>>>(A3, Wt[3], bias[3], hs + 3 * SPL, cs + 3 * SPL,
                                       nullptr,      Afc, 2048, 1024);
        k_fc<<<250, 256, 0, stream>>>(Afc, fcWt, fcb, logits, t);
        k_argmax<<<32, 256, 0, stream>>>(logits, t, tok, tokout);
    }
}
